// MERG_22204980920682
// MI455X (gfx1250) — compile-verified, matrix operand order checked by reading, not run on hardware
//
#include <hip/hip_runtime.h>
#include <hip/hip_bf16.h>
#include <cstdint>
#include <cstddef>

// ---------------------------------------------------------------------------
// Types for CDNA5 WMMA (wave32): v_wmma_f32_16x16x32_bf16
// ---------------------------------------------------------------------------
typedef __attribute__((ext_vector_type(16))) __bf16 v16bf;
typedef __attribute__((ext_vector_type(8)))  __bf16 v8bf;
typedef __attribute__((ext_vector_type(8)))  float  v8f;

static __device__ __forceinline__ v16bf make_frag(const __bf16* p) {
  // A/B 16-bit fragment: VGPR0..3 = K[h*8 .. h*8+7], VGPR4..7 = K[16+h*8 ..]
  // p = &tile[row][h*8]; both halves are 16B-aligned b128 LDS loads.
  union U { v16bf v; struct { v8bf lo, hi; } s; } u;
  u.s.lo = *(const v8bf*)(p);
  u.s.hi = *(const v8bf*)(p + 16);
  return u.v;
}

static __device__ __forceinline__ v8bf cvt8(const float4& a, const float4& b) {
  v8bf r;
  r[0] = (__bf16)a.x; r[1] = (__bf16)a.y; r[2] = (__bf16)a.z; r[3] = (__bf16)a.w;
  r[4] = (__bf16)b.x; r[5] = (__bf16)b.y; r[6] = (__bf16)b.z; r[7] = (__bf16)b.w;
  return r;
}

// Async DMA: global -> LDS, 16 bytes per lane, tracked by ASYNCcnt.
// The compiler's __syncthreads() does NOT wait on ASYNCcnt, so every use is
// followed by an explicit s_wait_asynccnt fence before the barrier.
static __device__ __forceinline__ void async_copy_b128(unsigned lds_addr,
                                                       const void* gaddr) {
  asm volatile("global_load_async_to_lds_b128 %0, %1, off"
               :: "v"(lds_addr), "v"(gaddr) : "memory");
}
static __device__ __forceinline__ void async_fence() {
  asm volatile("s_wait_asynccnt 0x0" ::: "memory");
}

// ---------------------------------------------------------------------------
// Generic WMMA GEMM:  D = epi( (A @ B + bias + Cadd) * rowScale )
//   A: fp32 (optional row-gather + override + K-concat) or bf16 direct
//   B: pre-transposed bf16 weights Bt[n*K + k] = B[k][n]
// 256 threads = 8 waves; tile BM=128 x BN=64, BK=32; double-buffered LDS.
// bf16 tiles are DMA'd straight into LDS with async-to-LDS b128 ops.
// ---------------------------------------------------------------------------
constexpr int BM = 128, BN = 64, BK = 32, LDT = BK + 8;

template<bool ABF16, int EPI, bool OUTBF>
__global__ __launch_bounds__(256)
void gemm_wmma(const void* __restrict__ Ap, int lda,
               const int* __restrict__ ridx, const float* __restrict__ Aover, int nover,
               const float* __restrict__ Acat, int ldcat, int kSplit,
               const __bf16* __restrict__ Bt,
               const float* __restrict__ bias,
               const float* __restrict__ rowScale,
               const float* __restrict__ Cadd, int ldadd,
               void* __restrict__ Dp, int ldd,
               int Mtot, int Ntot, int Ktot)
{
  __shared__ __bf16 sA[2][BM * LDT];
  __shared__ __bf16 sB[2][BN * LDT];
  const int tid  = threadIdx.x;
  const int lane = tid & 31;
  const int wave = tid >> 5;
  const int m0 = blockIdx.x * BM;
  const int n0 = blockIdx.y * BN;
  const int wm = (wave & 3) * 32;
  const int wn = (wave >> 2) * 32;
  const int fm = lane & 15;
  const int fh = lane >> 4;

  // staging coordinates (fixed per thread)
  const int ar  = tid >> 1;            // A: 2 threads/row, 16 elems each
  const int ac0 = (tid & 1) * 16;
  int agr = m0 + ar; if (agr > Mtot - 1) agr = Mtot - 1;   // clamp, EXEC stays full
  const int br  = tid >> 2;            // B: 4 threads/row, 8 elems each
  const int bc0 = (tid & 3) * 8;

  auto loadA = [&](int bufi, int k0) {
    __bf16* dst = &sA[bufi][ar * LDT + ac0];
    if constexpr (ABF16) {
      const __bf16* src = (const __bf16*)Ap + (size_t)agr * lda + (k0 + ac0);
      unsigned d0 = (unsigned)(uintptr_t)dst;
      async_copy_b128(d0,      (const void*)src);
      async_copy_b128(d0 + 16, (const void*)(src + 8));
    } else {
      int gk = k0 + ac0;
      const float* src;
      if (Acat != nullptr && gk >= kSplit) {
        src = Acat + (size_t)agr * ldcat + (gk - kSplit);        // concat half
      } else {
        int rr = (ridx != nullptr) ? ridx[agr] : agr;            // gather
        const float* base = (Aover != nullptr && rr < nover) ? Aover
                                                             : (const float*)Ap;
        src = base + (size_t)rr * lda + gk;
      }
      float4 f0 = *(const float4*)(src);
      float4 f1 = *(const float4*)(src + 4);
      float4 f2 = *(const float4*)(src + 8);
      float4 f3 = *(const float4*)(src + 12);
      *(v8bf*)(dst)     = cvt8(f0, f1);
      *(v8bf*)(dst + 8) = cvt8(f2, f3);
    }
  };
  auto loadB = [&](int bufi, int k0) {
    const __bf16* src = Bt + (size_t)(n0 + br) * Ktot + (k0 + bc0);
    __builtin_prefetch((const void*)(src + 2 * BK), 0, 1);  // global_prefetch_b8
    async_copy_b128((unsigned)(uintptr_t)&sB[bufi][br * LDT + bc0],
                    (const void*)src);
  };

  v8f acc[2][2] = {};
  loadA(0, 0);
  loadB(0, 0);
  async_fence();
  __syncthreads();

  const int nk = Ktot / BK;
  for (int t = 0; t < nk; ++t) {
    const int cur = t & 1;
    if (t + 1 < nk) {               // stage next tile while computing current
      loadA(cur ^ 1, (t + 1) * BK);
      loadB(cur ^ 1, (t + 1) * BK);
    }
    v16bf af[2], bf2[2];
    #pragma unroll
    for (int mi = 0; mi < 2; ++mi)
      af[mi] = make_frag(&sA[cur][(wm + mi * 16 + fm) * LDT + fh * 8]);
    #pragma unroll
    for (int ni = 0; ni < 2; ++ni)
      bf2[ni] = make_frag(&sB[cur][(wn + ni * 16 + fm) * LDT + fh * 8]);
    #pragma unroll
    for (int mi = 0; mi < 2; ++mi)
      #pragma unroll
      for (int ni = 0; ni < 2; ++ni)
        acc[mi][ni] = __builtin_amdgcn_wmma_f32_16x16x32_bf16(
            false, af[mi], false, bf2[ni], (short)0, acc[mi][ni], false, false);
    async_fence();                  // async DMAs for next tile must land
    __syncthreads();
  }

  // ---- epilogue (C layout: lane -> col, vgpr i -> row fh*8+i) ----
  const bool partial = (m0 + BM) > Mtot;   // block-uniform
  #pragma unroll
  for (int ni = 0; ni < 2; ++ni) {
    const int col = n0 + wn + ni * 16 + fm;
    const float bv = (bias != nullptr) ? bias[col] : 0.f;
    #pragma unroll
    for (int mi = 0; mi < 2; ++mi) {
      const int rbase = m0 + wm + mi * 16 + fh * 8;
      #pragma unroll
      for (int i = 0; i < 8; ++i) {
        const int row = rbase + i;
        if (!partial || row < Mtot) {
          float v = acc[mi][ni][i] + bv;
          if (Cadd     != nullptr) v += Cadd[(size_t)row * ldadd + col];
          if (rowScale != nullptr) v *= rowScale[row];
          if constexpr (EPI == 1) v = (v > 0.f ? v : __expf(v) - 1.f) + 1.f; // elu+1
          if constexpr (EPI == 2) v = v > 0.f ? v : 0.f;                     // relu
          if constexpr (OUTBF) ((__bf16*)Dp)[(size_t)row * ldd + col] = (__bf16)v;
          else                 ((float*)Dp)[(size_t)row * ldd + col] = v;
        }
      }
    }
  }
}

// ---------------------------------------------------------------------------
// KV = K^T @ V  (split-K over S with f32 atomic accumulation into KV[D][D]).
// Operands bf16 [S,D]; unconditional clamped b128 loads, zero-select for the
// S tail, transposed scatter into LDS, double-buffered.
// ---------------------------------------------------------------------------
__global__ __launch_bounds__(256)
void gemm_KtV(const __bf16* __restrict__ Kb, const __bf16* __restrict__ Vb,
              int S, int D, int SCHUNK, float* __restrict__ KV)
{
  __shared__ __bf16 sA[2][BM * LDT];   // sA[d1][s]
  __shared__ __bf16 sB[2][BN * LDT];   // sB[d2][s]
  const int tid  = threadIdx.x;
  const int lane = tid & 31;
  const int wave = tid >> 5;
  const int m0 = blockIdx.x * BM;
  const int n0 = blockIdx.y * BN;
  const int wm = (wave & 3) * 32;
  const int wn = (wave >> 2) * 32;
  const int fm = lane & 15;
  const int fh = lane >> 4;

  const int sr  = tid >> 3;            // 0..31  (s within tile)
  const int ac0 = (tid & 7) * 16;      // 16 d-elems of K per thread
  const int bc0 = (tid & 7) * 8;       // 8 d-elems of V per thread

  auto loadTiles = [&](int bufi, int s0) {
    int gs = s0 + sr;
    const bool ok = gs < S;
    if (!ok) gs = S - 1;               // clamped unconditional loads
    v8bf a0 = *(const v8bf*)(Kb + (size_t)gs * D + m0 + ac0);
    v8bf a1 = *(const v8bf*)(Kb + (size_t)gs * D + m0 + ac0 + 8);
    v8bf b0 = *(const v8bf*)(Vb + (size_t)gs * D + n0 + bc0);
    if (!ok) { v8bf z = {}; a0 = z; a1 = z; b0 = z; }
    #pragma unroll
    for (int i = 0; i < 8; ++i) {
      sA[bufi][(ac0 + i)     * LDT + sr] = a0[i];
      sA[bufi][(ac0 + 8 + i) * LDT + sr] = a1[i];
      sB[bufi][(bc0 + i)     * LDT + sr] = b0[i];
    }
  };

  v8f acc[2][2] = {};
  const int s_begin = blockIdx.z * SCHUNK;
  int s_last = s_begin + SCHUNK; if (s_last > S) s_last = S;
  const int nsteps = (s_last - s_begin + 31) / 32;

  loadTiles(0, s_begin);
  __syncthreads();
  for (int t = 0; t < nsteps; ++t) {
    const int cur = t & 1;
    if (t + 1 < nsteps) loadTiles(cur ^ 1, s_begin + (t + 1) * 32);
    v16bf af[2], bf2[2];
    #pragma unroll
    for (int mi = 0; mi < 2; ++mi)
      af[mi] = make_frag(&sA[cur][(wm + mi * 16 + fm) * LDT + fh * 8]);
    #pragma unroll
    for (int ni = 0; ni < 2; ++ni)
      bf2[ni] = make_frag(&sB[cur][(wn + ni * 16 + fm) * LDT + fh * 8]);
    #pragma unroll
    for (int mi = 0; mi < 2; ++mi)
      #pragma unroll
      for (int ni = 0; ni < 2; ++ni)
        acc[mi][ni] = __builtin_amdgcn_wmma_f32_16x16x32_bf16(
            false, af[mi], false, bf2[ni], (short)0, acc[mi][ni], false, false);
    __syncthreads();
  }

  #pragma unroll
  for (int mi = 0; mi < 2; ++mi)
    #pragma unroll
    for (int ni = 0; ni < 2; ++ni)
      #pragma unroll
      for (int i = 0; i < 8; ++i) {
        int row = m0 + wm + mi * 16 + fh * 8 + i;
        int col = n0 + wn + ni * 16 + fm;
        atomicAdd(&KV[(size_t)row * D + col], acc[mi][ni][i]);
      }
}

// ---------------------------------------------------------------------------
// Helper kernels
// ---------------------------------------------------------------------------
__global__ void colsum_bf16(const __bf16* __restrict__ Kb, int S, int D,
                            float* __restrict__ out)
{
  int col = threadIdx.x;                  // blockDim.x == D == 256
  float s = 0.f;
  for (int r = blockIdx.x; r < S; r += gridDim.x)
    s += (float)Kb[(size_t)r * D + col];
  atomicAdd(&out[col], s);
}

__global__ void compute_recipZ(const __bf16* __restrict__ Q,
                               const float* __restrict__ ksum,
                               int L, int D, float* __restrict__ recipZ)
{
  int row  = blockIdx.x * (blockDim.x >> 5) + (threadIdx.x >> 5);
  int lane = threadIdx.x & 31;
  if (row >= L) return;
  const __bf16* q = Q + (size_t)row * D;
  float s = 0.f;
  for (int c = lane; c < D; c += 32) s += (float)q[c] * ksum[c];
  #pragma unroll
  for (int o = 16; o > 0; o >>= 1) s += __shfl_xor(s, o, 32);
  if (lane == 0) recipZ[row] = 1.0f / (s + 1e-6f);
}

// LayerNorm over last dim (D), optional residual add (with gather/override),
// optional bf16 shadow output. One wave per row.
__global__ void layernorm_rows(const float* __restrict__ X, int L, int D,
                               const float* __restrict__ g, const float* __restrict__ b,
                               const float* __restrict__ Rprim, const float* __restrict__ Rover,
                               const int* __restrict__ Rridx, int Rnover,
                               float* __restrict__ Y, __bf16* __restrict__ Ybf)
{
  int row  = blockIdx.x * (blockDim.x >> 5) + (threadIdx.x >> 5);
  int lane = threadIdx.x & 31;
  if (row >= L) return;
  const float* x = X + (size_t)row * D;
  float s = 0.f, s2 = 0.f;
  for (int c = lane; c < D; c += 32) { float v = x[c]; s += v; s2 += v * v; }
  #pragma unroll
  for (int o = 16; o > 0; o >>= 1) { s += __shfl_xor(s, o, 32); s2 += __shfl_xor(s2, o, 32); }
  float mu   = s / D;
  float var  = s2 / D - mu * mu;
  float rstd = rsqrtf(var + 1e-5f);
  const float* r = nullptr;
  if (Rprim != nullptr) {
    int rr = (Rridx != nullptr) ? Rridx[row] : row;
    const float* base = (Rover != nullptr && rr < Rnover) ? Rover : Rprim;
    r = base + (size_t)rr * D;
  }
  for (int c = lane; c < D; c += 32) {
    float v = (x[c] - mu) * rstd * g[c] + b[c];
    if (r != nullptr) v += r[c];
    Y[(size_t)row * D + c] = v;
    if (Ybf != nullptr) Ybf[(size_t)row * D + c] = (__bf16)v;
  }
}

__global__ void bn_stats(const float* __restrict__ X, int Mrows, int D,
                         float* __restrict__ sums, float* __restrict__ sumsq)
{
  int col = threadIdx.x;                  // blockDim.x == D
  float s = 0.f, s2 = 0.f;
  for (int r = blockIdx.x; r < Mrows; r += gridDim.x) {
    float v = X[(size_t)r * D + col]; s += v; s2 += v * v;
  }
  atomicAdd(&sums[col], s);
  atomicAdd(&sumsq[col], s2);
}

__global__ void bn_apply(const float* __restrict__ X, size_t total, int D, int Mrows,
                         const float* __restrict__ sums, const float* __restrict__ sumsq,
                         const float* __restrict__ g, const float* __restrict__ b,
                         int do_relu, float* __restrict__ Y)
{
  size_t i = (size_t)blockIdx.x * blockDim.x + threadIdx.x;
  if (i >= total) return;
  int col   = (int)(i % D);
  float mu  = sums[col] / (float)Mrows;
  float var = sumsq[col] / (float)Mrows - mu * mu;
  float v   = (X[i] - mu) * rsqrtf(var + 1e-5f) * g[col] + b[col];
  if (do_relu) v = v > 0.f ? v : 0.f;
  Y[i] = v;
}

// Conv1d(2->1, k=3, pad=1) over gathered edge pairs, bf16 output for GEMM.
__global__ void edge_conv(const float* __restrict__ H,
                          const int* __restrict__ gsrc, const int* __restrict__ gdst,
                          const float* __restrict__ w, const float* __restrict__ cb,
                          int E, int D, __bf16* __restrict__ out)
{
  int e = blockIdx.x;
  int j = threadIdx.x;                    // blockDim.x == D
  if (e >= E) return;
  const float* s = H + (size_t)gsrc[e] * D;
  const float* d = H + (size_t)gdst[e] * D;
  float sl = (j > 0)     ? s[j - 1] : 0.f;
  float sr = (j < D - 1) ? s[j + 1] : 0.f;
  float dl = (j > 0)     ? d[j - 1] : 0.f;
  float dr = (j < D - 1) ? d[j + 1] : 0.f;
  float acc = cb[0] + w[0] * sl + w[1] * s[j] + w[2] * sr
                    + w[3] * dl + w[4] * d[j] + w[5] * dr;
  out[(size_t)e * D + j] = (__bf16)acc;
}

// Wt[n*K + k] = (bf16) W[k*N + n]
__global__ void convert_transpose(const float* __restrict__ W, int K, int N,
                                  __bf16* __restrict__ Wt)
{
  size_t total = (size_t)K * N;
  for (size_t i = (size_t)blockIdx.x * blockDim.x + threadIdx.x;
       i < total; i += (size_t)gridDim.x * blockDim.x) {
    int k = (int)(i / N), n = (int)(i % N);
    Wt[(size_t)n * K + k] = (__bf16)W[i];
  }
}

// ---------------------------------------------------------------------------
// Host-side encoder orchestration
// ---------------------------------------------------------------------------
struct EncW {
  const __bf16 *Wqt, *Wkt, *Wvt, *Wmt, *mw1t, *mw2t;
  const float  *bm, *ln1g, *ln1b, *mb1, *mb2, *ln2g, *ln2b;
};
struct EncBufs {
  __bf16 *Q, *K, *V, *msg, *y, *KVt;
  float  *t, *msgln, *KV, *ksum, *recipZ;
};

static inline int cdiv(int a, int b) { return (a + b - 1) / b; }

static void run_encoder(hipStream_t stream,
    const float* Lp, const float* Lo, const int* Lr, int Lnov, int L,
    const float* Sp, const float* So, const int* Sr, int Snov, int S,
    const EncW& w, const EncBufs& bf, float* out, __bf16* outbf)
{
  const int D = 256;
  dim3 blk(256);
  // Q = elu(x@Wq)+1 ; K = elu(src@Wk)+1 ; V = src@Wv        (bf16 out)
  gemm_wmma<false,1,true><<<dim3(cdiv(L,BM),D/BN), blk, 0, stream>>>(
      Lp, D, Lr, Lo, Lnov, nullptr,0,0, w.Wqt, nullptr,nullptr,nullptr,0,
      bf.Q, D, L, D, D);
  gemm_wmma<false,1,true><<<dim3(cdiv(S,BM),D/BN), blk, 0, stream>>>(
      Sp, D, Sr, So, Snov, nullptr,0,0, w.Wkt, nullptr,nullptr,nullptr,0,
      bf.K, D, S, D, D);
  gemm_wmma<false,0,true><<<dim3(cdiv(S,BM),D/BN), blk, 0, stream>>>(
      Sp, D, Sr, So, Snov, nullptr,0,0, w.Wvt, nullptr,nullptr,nullptr,0,
      bf.V, D, S, D, D);
  // KV = K^T V (split-K atomics) ; ksum = colsum(K)
  (void)hipMemsetAsync(bf.KV,   0, (size_t)D * D * sizeof(float), stream);
  (void)hipMemsetAsync(bf.ksum, 0, (size_t)D * sizeof(float), stream);
  gemm_KtV<<<dim3(D/BM, D/BN, cdiv(S,512)), blk, 0, stream>>>(bf.K, bf.V, S, D, 512, bf.KV);
  colsum_bf16<<<dim3(512), blk, 0, stream>>>(bf.K, S, D, bf.ksum);
  convert_transpose<<<dim3(cdiv(D*D,256)), blk, 0, stream>>>(bf.KV, D, D, bf.KVt);
  // Z and msg = (Q@KV)/Z        (bf16 out)
  compute_recipZ<<<dim3(cdiv(L,8)), blk, 0, stream>>>(bf.Q, bf.ksum, L, D, bf.recipZ);
  gemm_wmma<true,0,true><<<dim3(cdiv(L,BM),D/BN), blk, 0, stream>>>(
      bf.Q, D, nullptr,nullptr,0, nullptr,0,0, bf.KVt, nullptr, bf.recipZ, nullptr,0,
      bf.msg, D, L, D, D);
  // t = msg@Wm + bm ; msgln = LN(t)
  gemm_wmma<true,0,false><<<dim3(cdiv(L,BM),D/BN), blk, 0, stream>>>(
      bf.msg, D, nullptr,nullptr,0, nullptr,0,0, w.Wmt, w.bm, nullptr,nullptr,0,
      bf.t, D, L, D, D);
  layernorm_rows<<<dim3(cdiv(L,8)), blk, 0, stream>>>(
      bf.t, L, D, w.ln1g, w.ln1b, nullptr,nullptr,nullptr,0, bf.msgln, (__bf16*)nullptr);
  // y = relu(concat(x, msgln) @ mw1 + mb1)   (bf16, K=512, N=512)
  gemm_wmma<false,2,true><<<dim3(cdiv(L,BM),(2*D)/BN), blk, 0, stream>>>(
      Lp, D, Lr, Lo, Lnov, bf.msgln, D, D, w.mw1t, w.mb1, nullptr,nullptr,0,
      bf.y, 2*D, L, 2*D, 2*D);
  // y2 = y @ mw2 + mb2  (reuse bf.t)
  gemm_wmma<true,0,false><<<dim3(cdiv(L,BM),D/BN), blk, 0, stream>>>(
      bf.y, 2*D, nullptr,nullptr,0, nullptr,0,0, w.mw2t, w.mb2, nullptr,nullptr,0,
      bf.t, D, L, D, 2*D);
  // out = x + LN(y2)
  layernorm_rows<<<dim3(cdiv(L,8)), blk, 0, stream>>>(
      bf.t, L, D, w.ln2g, w.ln2b, Lp, Lo, Lr, Lnov, out, outbf);
}

// ---------------------------------------------------------------------------
// Entry point
// ---------------------------------------------------------------------------
extern "C" void kernel_launch(void* const* d_in, const int* in_sizes, int n_in,
                              void* d_out, int out_size, void* d_ws, size_t ws_size,
                              hipStream_t stream)
{
  (void)in_sizes; (void)n_in; (void)out_size; (void)ws_size;
  const int N = 20000, E = 200000, ES = 100000, D = 256;

  const float* h_nodes = (const float*)d_in[0];
  const float* s_u     = (const float*)d_in[1];
  const int*   g_src   = (const int*)d_in[2];
  const int*   g_dst   = (const int*)d_in[3];
  const int*   s_src   = (const int*)d_in[4];
  const int*   s_dst   = (const int*)d_in[5];
  const float* conv_w  = (const float*)d_in[6];
  const float* conv_b  = (const float*)d_in[7];
  const float* W2      = (const float*)d_in[8];
  const float* b2      = (const float*)d_in[9];
  const float* W4      = (const float*)d_in[10];
  const float* b4      = (const float*)d_in[11];
  const float* bn_g    = (const float*)d_in[12];
  const float* bn_b    = (const float*)d_in[13];
  const float* Wq      = (const float*)d_in[14];
  const float* Wk      = (const float*)d_in[15];
  const float* Wv      = (const float*)d_in[16];
  const float* Wm      = (const float*)d_in[17];
  const float* bm      = (const float*)d_in[18];
  const float* ln1_g   = (const float*)d_in[19];
  const float* ln1_b   = (const float*)d_in[20];
  const float* mw1     = (const float*)d_in[21];
  const float* mb1     = (const float*)d_in[22];
  const float* mw2     = (const float*)d_in[23];
  const float* mb2     = (const float*)d_in[24];
  const float* ln2_g   = (const float*)d_in[25];
  const float* ln2_b   = (const float*)d_in[26];
  float* out = (float*)d_out;

  // ---- workspace bump allocator ----
  char* wsp = (char*)d_ws;
  auto alloc = [&](size_t bytes) -> void* {
    void* r = wsp; wsp += (bytes + 255) & ~(size_t)255; return r;
  };
  const size_t ED  = (size_t)E * D;
  __bf16* Wqt  = (__bf16*)alloc((size_t)D * D * 2);
  __bf16* Wkt  = (__bf16*)alloc((size_t)D * D * 2);
  __bf16* Wvt  = (__bf16*)alloc((size_t)D * D * 2);
  __bf16* Wmt  = (__bf16*)alloc((size_t)D * D * 2);
  __bf16* W2t  = (__bf16*)alloc((size_t)D * D * 2);
  __bf16* W4t  = (__bf16*)alloc((size_t)D * D * 2);
  __bf16* mw1t = (__bf16*)alloc((size_t)(2*D) * (2*D) * 2);
  __bf16* mw2t = (__bf16*)alloc((size_t)(2*D) * D * 2);
  EncBufs bf;
  bf.Q      = (__bf16*)alloc(ED * 2);
  bf.K      = (__bf16*)alloc(ED * 2);
  bf.V      = (__bf16*)alloc(ED * 2);
  bf.msg    = (__bf16*)alloc(ED * 2);
  bf.y      = (__bf16*)alloc(ED * 2 * 2);          // [E, 2D] bf16
  bf.KVt    = (__bf16*)alloc((size_t)D * D * 2);
  bf.t      = (float*)alloc(ED * 4);
  bf.msgln  = (float*)alloc(ED * 4);
  bf.KV     = (float*)alloc((size_t)D * D * 4);
  bf.ksum   = (float*)alloc((size_t)D * 4);
  bf.recipZ = (float*)alloc((size_t)E * 4);
  float* xbuf   = (float*)alloc((size_t)N * D * 4);   // enc1 output, persists
  float* encOut = (float*)alloc(ED * 4);              // enc2 / enc3 output
  __bf16* encBf = (__bf16*)alloc(ED * 2);             // bf16 shadow of enc2 out
  float* lrbuf  = (float*)alloc(ED * 4);              // pre-BN lr_e
  float* bnsum  = (float*)alloc((size_t)D * 4);
  float* bnsq   = (float*)alloc((size_t)D * 4);

  dim3 blk(256);

  // ---- convert all weights to bf16, transposed for the B-fragment path ----
  convert_transpose<<<dim3(256), blk, 0, stream>>>(Wq,  D,   D,   Wqt);
  convert_transpose<<<dim3(256), blk, 0, stream>>>(Wk,  D,   D,   Wkt);
  convert_transpose<<<dim3(256), blk, 0, stream>>>(Wv,  D,   D,   Wvt);
  convert_transpose<<<dim3(256), blk, 0, stream>>>(Wm,  D,   D,   Wmt);
  convert_transpose<<<dim3(256), blk, 0, stream>>>(W2,  D,   D,   W2t);
  convert_transpose<<<dim3(256), blk, 0, stream>>>(W4,  D,   D,   W4t);
  convert_transpose<<<dim3(1024), blk, 0, stream>>>(mw1, 2*D, 2*D, mw1t);
  convert_transpose<<<dim3(512),  blk, 0, stream>>>(mw2, 2*D, D,   mw2t);

  // ---- local edge features: conv (into bf.Q as scratch) then @W2+b2 ----
  edge_conv<<<dim3(E), blk, 0, stream>>>(h_nodes, g_src, g_dst, conv_w, conv_b, E, D, bf.Q);
  gemm_wmma<true,0,false><<<dim3(cdiv(E,BM), D/BN), blk, 0, stream>>>(
      bf.Q, D, nullptr,nullptr,0, nullptr,0,0, W2t, b2, nullptr,nullptr,0,
      lrbuf, D, E, D, D);

  EncW w = { Wqt, Wkt, Wvt, Wmt, mw1t, mw2t,
             bm, ln1_g, ln1_b, mb1, mb2, ln2_g, ln2_b };

  // ---- enc1: x = enc(h_nodes[:N], s_u) ----
  run_encoder(stream, h_nodes, nullptr, nullptr, 0, N,
                      s_u,     nullptr, nullptr, 0, N,
              w, bf, xbuf, nullptr);

  // ---- enc2: edge = enc(h_n[g_src], h_n[g_dst]); h_n = rows<N overridden by x
  run_encoder(stream, h_nodes, xbuf, g_src, N, E,
                      h_nodes, xbuf, g_dst, N, E,
              w, bf, encOut, encBf);

  // ---- lr_e(pre-BN) = edge@W4 + b4 + lr_e_local ----
  gemm_wmma<true,0,false><<<dim3(cdiv(E,BM), D/BN), blk, 0, stream>>>(
      encBf, D, nullptr,nullptr,0, nullptr,0,0, W4t, b4, nullptr, lrbuf, D,
      lrbuf, D, E, D, D);

  // ---- enc3: social = enc(x[s_src], x[s_dst]) ----
  run_encoder(stream, xbuf, nullptr, s_src, 0, ES,
                      xbuf, nullptr, s_dst, 0, ES,
              w, bf, encOut, nullptr);

  // ---- BatchNorm(train) + relu on lr_e -> out[0 .. E*D) ----
  (void)hipMemsetAsync(bnsum, 0, (size_t)D * 4, stream);
  (void)hipMemsetAsync(bnsq,  0, (size_t)D * 4, stream);
  bn_stats<<<dim3(1024), blk, 0, stream>>>(lrbuf, E, D, bnsum, bnsq);
  bn_apply<<<dim3((int)((ED + 255) / 256)), blk, 0, stream>>>(
      lrbuf, ED, D, E, bnsum, bnsq, bn_g, bn_b, 1, out);

  // ---- BatchNorm(train) on social_edge -> out[E*D .. E*D+ES*D) ----
  const size_t SD = (size_t)ES * D;
  (void)hipMemsetAsync(bnsum, 0, (size_t)D * 4, stream);
  (void)hipMemsetAsync(bnsq,  0, (size_t)D * 4, stream);
  bn_stats<<<dim3(1024), blk, 0, stream>>>(encOut, ES, D, bnsum, bnsq);
  bn_apply<<<dim3((int)((SD + 255) / 256)), blk, 0, stream>>>(
      encOut, SD, D, ES, bnsum, bnsq, bn_g, bn_b, 0, out + ED);
}
